// Attention_45363444580982
// MI455X (gfx1250) — matrix operand order repaired and checked by reading, not compiled, not run on hardware
//
#include <hip/hip_runtime.h>
#include <hip/hip_bf16.h>
#include <math.h>

// Problem constants (match the reference)
constexpr int Bb = 64;
constexpr int Ss = 4096;
constexpr int Hh = 256;

typedef __attribute__((ext_vector_type(16))) __bf16         v16bf;
typedef __attribute__((ext_vector_type(8)))  float          v8f;
typedef __attribute__((ext_vector_type(8)))  unsigned int   v8u;

static __device__ __forceinline__ unsigned short f2bf_u(float f) {
    __bf16 h = (__bf16)f;                       // RNE convert, hardware v_cvt
    return __builtin_bit_cast(unsigned short, h);
}

// ---------------------------------------------------------------------------
// Prep 1: convert W2 = attn_w[:, H:] (f32, row-major [n][k]) to bf16 [n][k]
// ---------------------------------------------------------------------------
__global__ void w2_convert_kernel(const float* __restrict__ attn_w,
                                  unsigned short* __restrict__ w2b) {
    int t = threadIdx.x;
    int base = blockIdx.x * 1024;               // 64 blocks * 1024 = 65536
    #pragma unroll
    for (int i = 0; i < 4; ++i) {
        int g = base + i * 256 + t;             // 0 .. 65535
        int n = g >> 8;                         // output channel
        int k = g & 255;                        // input h
        w2b[g] = f2bf_u(attn_w[(size_t)n * (2 * Hh) + Hh + k]);
    }
}

// ---------------------------------------------------------------------------
// Prep 2: h_proj[b][k] = sum_h hidden[b][h] * attn_w[k][h] + attn_b[k]
// ---------------------------------------------------------------------------
__global__ void hproj_kernel(const float* __restrict__ hidden,
                             const float* __restrict__ attn_w,
                             const float* __restrict__ attn_b,
                             float* __restrict__ hproj) {
    __shared__ float sh[Hh];
    int b = blockIdx.x, t = threadIdx.x;
    sh[t] = hidden[b * Hh + t];
    __syncthreads();
    const float* wr = attn_w + (size_t)t * (2 * Hh);
    float acc = attn_b[t];
    #pragma unroll 8
    for (int h = 0; h < Hh; ++h) acc = fmaf(sh[h], wr[h], acc);
    hproj[b * Hh + t] = acc;
}

// ---------------------------------------------------------------------------
// Main: scores[b][s] = v . tanh(outputs[b,s,:] @ W2^T + h_proj[b,:])
// Block: 256 threads (8 wave32), covers 64 s-rows x all 256 channels.
// A tile staged in LDS as bf16; GEMM via v_wmma_f32_16x16x32_bf16.
// ---------------------------------------------------------------------------
constexpr int RS = 264;                          // LDS row stride (ushorts), padded

__global__ __launch_bounds__(256) void scores_kernel(
        const float* __restrict__ outs,
        const unsigned short* __restrict__ w2b,
        const float* __restrict__ hproj,
        const float* __restrict__ vw,
        float* __restrict__ scores) {
    __shared__ unsigned short sA[64 * RS];       // 33792 B: 64 rows x 256 K (bf16)
    __shared__ float sScore[64];

    const int t    = threadIdx.x;
    const int lane = t & 31;
    const int wave = t >> 5;
    const int b    = blockIdx.y;
    const int s0   = blockIdx.x * 64;

    if (t < 64) sScore[t] = 0.0f;

    // ---- stage A tile (64 x 256 f32 -> bf16 in LDS), float4-wide ----
    {
        const float* gbase = outs + ((size_t)b * Ss + s0) * Hh;
        #pragma unroll
        for (int it = 0; it < 16; ++it) {
            int g   = it * 1024 + t * 4;         // element index in 64x256 tile
            int row = g >> 8;
            int col = g & 255;
            float4 f = *(const float4*)(gbase + (size_t)row * Hh + col);
            unsigned int u0 = (unsigned int)f2bf_u(f.x) | ((unsigned int)f2bf_u(f.y) << 16);
            unsigned int u1 = (unsigned int)f2bf_u(f.z) | ((unsigned int)f2bf_u(f.w) << 16);
            *(uint2*)(&sA[row * RS + col]) = make_uint2(u0, u1);
        }
    }
    __syncthreads();

    // ---- WMMA GEMM: this wave owns N-tiles {2*wave, 2*wave+1}, M-tiles 0..3 ----
    const int half = lane >> 4;                  // K-half select
    const int mr   = lane & 15;                  // row within M-tile / n within N-tile

    v8f acc[2][4];
    #pragma unroll
    for (int j = 0; j < 2; ++j)
        #pragma unroll
        for (int mt = 0; mt < 4; ++mt)
            acc[j][mt] = (v8f){0.f, 0.f, 0.f, 0.f, 0.f, 0.f, 0.f, 0.f};

    for (int ks = 0; ks < 8; ++ks) {
        const int k0 = ks * 32;

        // A fragments from LDS (ISA 16-bit A layout: lanes 0-15 K={0..7,16..23})
        v16bf afr[4];
        #pragma unroll
        for (int mt = 0; mt < 4; ++mt) {
            const unsigned short* rp = &sA[(mt * 16 + mr) * RS + k0 + half * 8];
            uint4 lo = *(const uint4*)(rp);
            uint4 hi = *(const uint4*)(rp + 16);
            v8u packed;
            packed[0] = lo.x; packed[1] = lo.y; packed[2] = lo.z; packed[3] = lo.w;
            packed[4] = hi.x; packed[5] = hi.y; packed[6] = hi.z; packed[7] = hi.w;
            afr[mt] = __builtin_bit_cast(v16bf, packed);
        }

        #pragma unroll
        for (int j = 0; j < 2; ++j) {
            const int n0 = (wave * 2 + j) * 16;
            // B fragment from global bf16 W2 (lanes 0-15 K={0..7,16..23}, 16-31 K={8..15,24..31})
            const unsigned short* wp = w2b + (size_t)(n0 + mr) * Hh + k0 + half * 8;
            uint4 q0 = *(const uint4*)(wp);
            uint4 q1 = *(const uint4*)(wp + 16);
            v8u packed;
            packed[0] = q0.x; packed[1] = q0.y; packed[2] = q0.z; packed[3] = q0.w;
            packed[4] = q1.x; packed[5] = q1.y; packed[6] = q1.z; packed[7] = q1.w;
            v16bf bfr = __builtin_bit_cast(v16bf, packed);

            #pragma unroll
            for (int mt = 0; mt < 4; ++mt) {
                acc[j][mt] = __builtin_amdgcn_wmma_f32_16x16x32_bf16(
                    false, afr[mt], false, bfr, (short)0, acc[j][mt], false, false);
            }
        }
    }

    __syncthreads();   // sScore zero-init visible; also A staging done

    // ---- epilogue: tanh, dot with v_w, reduce to per-row scores ----
    float rowacc[4][8];
    #pragma unroll
    for (int mt = 0; mt < 4; ++mt)
        #pragma unroll
        for (int r = 0; r < 8; ++r) rowacc[mt][r] = 0.0f;

    #pragma unroll
    for (int j = 0; j < 2; ++j) {
        const int n = (wave * 2 + j) * 16 + mr;
        const float hp = hproj[b * Hh + n];
        const float vv = vw[n];
        #pragma unroll
        for (int mt = 0; mt < 4; ++mt)
            #pragma unroll
            for (int r = 0; r < 8; ++r)
                rowacc[mt][r] = fmaf(vv, tanhf(acc[j][mt][r] + hp), rowacc[mt][r]);
    }

    #pragma unroll
    for (int mt = 0; mt < 4; ++mt) {
        #pragma unroll
        for (int r = 0; r < 8; ++r) {
            float v = rowacc[mt][r];
            v += __shfl_xor(v, 1, 32);
            v += __shfl_xor(v, 2, 32);
            v += __shfl_xor(v, 4, 32);
            v += __shfl_xor(v, 8, 32);           // sum over 16 lanes (one K-half group)
            if (mr == 0)
                atomicAdd(&sScore[mt * 16 + half * 8 + r], v);  // ds_add_f32
        }
    }
    __syncthreads();

    if (t < 64) scores[(size_t)b * Ss + s0 + t] = sScore[t];
}

// ---------------------------------------------------------------------------
// Softmax over S per batch; also zero-init context region of d_out.
// ---------------------------------------------------------------------------
__global__ void softmax_kernel(const float* __restrict__ scores,
                               float* __restrict__ out) {
    __shared__ float red[256];
    const int b = blockIdx.x, t = threadIdx.x;
    float* alpha   = out + Bb * Hh;
    float* context = out;

    float v[16];
    float mx = -3.402823e38f;
    #pragma unroll
    for (int i = 0; i < 16; ++i) {
        v[i] = scores[(size_t)b * Ss + i * 256 + t];
        mx = fmaxf(mx, v[i]);
    }
    red[t] = mx; __syncthreads();
    for (int st = 128; st > 0; st >>= 1) {
        if (t < st) red[t] = fmaxf(red[t], red[t + st]);
        __syncthreads();
    }
    mx = red[0]; __syncthreads();

    float sum = 0.0f;
    #pragma unroll
    for (int i = 0; i < 16; ++i) { v[i] = __expf(v[i] - mx); sum += v[i]; }
    red[t] = sum; __syncthreads();
    for (int st = 128; st > 0; st >>= 1) {
        if (t < st) red[t] += red[t + st];
        __syncthreads();
    }
    const float inv = 1.0f / red[0];
    #pragma unroll
    for (int i = 0; i < 16; ++i)
        alpha[(size_t)b * Ss + i * 256 + t] = v[i] * inv;

    context[b * Hh + t] = 0.0f;
}

// ---------------------------------------------------------------------------
// context[b][h] = sum_s alpha[b][s] * outputs[b][s][h]  (streaming, HBM-bound)
// ---------------------------------------------------------------------------
__global__ void context_kernel(const float* __restrict__ outs,
                               float* __restrict__ out) {
    __shared__ float al[256];
    const int b = blockIdx.y, c = blockIdx.x, t = threadIdx.x;
    const float* alpha = out + Bb * Hh;
    al[t] = alpha[(size_t)b * Ss + c * 256 + t];
    __syncthreads();

    const float* base = outs + ((size_t)b * Ss + (size_t)c * 256) * Hh + t;
    float acc = 0.0f;
    #pragma unroll 8
    for (int i = 0; i < 256; ++i)
        acc = fmaf(al[i], base[(size_t)i * Hh], acc);

    unsafeAtomicAdd(out + b * Hh + t, acc);      // global_atomic_add_f32
}

// ---------------------------------------------------------------------------
extern "C" void kernel_launch(void* const* d_in, const int* in_sizes, int n_in,
                              void* d_out, int out_size, void* d_ws, size_t ws_size,
                              hipStream_t stream) {
    const float* outs       = (const float*)d_in[0];   // [B,S,H]
    const float* hidden_all = (const float*)d_in[1];   // [L,B,H]
    const float* attn_w     = (const float*)d_in[2];   // [H,2H]
    const float* attn_b     = (const float*)d_in[3];   // [H]
    const float* v_w        = (const float*)d_in[4];   // [H]
    float* out = (float*)d_out;                        // context [B,H] ++ alpha [B,S]

    char* ws = (char*)d_ws;
    unsigned short* w2b = (unsigned short*)ws;                     // 131072 B
    float* hproj  = (float*)(ws + 131072);                         //  65536 B
    float* scores = (float*)(ws + 131072 + 65536);                 // 1 MiB

    const float* hidden = hidden_all + (size_t)1 * Bb * Hh;        // last layer

    w2_convert_kernel<<<64, 256, 0, stream>>>(attn_w, w2b);
    hproj_kernel<<<Bb, 256, 0, stream>>>(hidden, attn_w, attn_b, hproj);
    scores_kernel<<<dim3(Ss / 64, Bb), 256, 0, stream>>>(outs, w2b, hproj, v_w, scores);
    softmax_kernel<<<Bb, 256, 0, stream>>>(scores, out);
    context_kernel<<<dim3(Ss / 256, Bb), 256, 0, stream>>>(outs, out);
}
